// SparseConv2d_79121887527369
// MI455X (gfx1250) — matrix operand order repaired and checked by reading, not compiled, not run on hardware
//
#include <hip/hip_runtime.h>
#include <hip/hip_bf16.h>
#include <stdint.h>

// ---------------- Types for WMMA fragments (wave32, gfx1250) ----------------
typedef __attribute__((ext_vector_type(16))) __bf16 v16bf;
typedef __attribute__((ext_vector_type(8)))  __bf16 v8bf;
typedef __attribute__((ext_vector_type(8)))  float  v8f;

// Problem constants
#define NB   4      // batch
#define CIN  256    // input channels
#define COUT 256    // output channels
#define CPF  32     // channels per filter
#define HH   56
#define WW   56
#define HP   58     // padded height (h+2)
#define WP   66     // padded width  (covers w in [0,63] + kw in [0,2])

// Workspace layout (bf16 elements):
//   xt : [NB][HP][WP][CIN]  = 4*58*66*256 = 3,919,872 elems (7,839,744 B)
//   wdt: [9][COUT][CIN]     = 589,824 elems (1,179,648 B)
#define XT_ELEMS ((size_t)NB * HP * WP * CIN)

// f32 -> bf16 round-to-nearest-even via bit ops (no dependence on __bf16 scalar cvt)
__device__ __forceinline__ unsigned short f2bf(float f) {
    unsigned u = __builtin_bit_cast(unsigned, f);
    u += 0x7FFFu + ((u >> 16) & 1u);
    return (unsigned short)(u >> 16);
}

// ---------------- Kernel 1: zero the padded-input region --------------------
__global__ void zero32_kernel(unsigned int* __restrict__ p, int n) {
    int i = blockIdx.x * 256 + threadIdx.x;
    if (i < n) p[i] = 0u;
}

// ---------------- Kernel 2: NCHW f32 -> padded NHWC bf16 (LDS transpose) ----
// grid: (NB*HH, CIN/32, 2), block: (32, 8)
__global__ void pack_x_kernel(const float* __restrict__ x,
                              unsigned short* __restrict__ xt) {
    int bh = blockIdx.x;
    int b  = bh / HH, h = bh % HH;
    int cb = blockIdx.y;          // channel block of 32
    int wb = blockIdx.z;          // w block of 32 (covers 0..55)
    int tx = threadIdx.x;         // 0..31
    int ty = threadIdx.y;         // 0..7

    __shared__ float tile[32][33];  // [w_local][c_local], padded stride

    int w = wb * 32 + tx;
    if (w < WW) {
        #pragma unroll
        for (int i = 0; i < 4; ++i) {
            int cl = ty + 8 * i;
            tile[tx][cl] =
                x[(((size_t)b * CIN + cb * 32 + cl) * HH + h) * WW + w];
        }
    }
    __syncthreads();
    #pragma unroll
    for (int i = 0; i < 4; ++i) {
        int wl = ty + 8 * i;
        int wo = wb * 32 + wl;
        if (wo < WW) {
            // +1 halo in both h and w (SAME padding, pad=1)
            xt[((size_t)(b * HP + h + 1) * WP + wo + 1) * CIN + cb * 32 + tx] =
                f2bf(tile[wl][tx]);
        }
    }
}

// ---------------- Kernel 3: densify sparse weights --------------------------
// wdt[t][o][ci] = sum over c where connections[o][c]==ci of weight[o][c][t]
// grid: COUT blocks, 256 threads (one per ci). No atomics needed.
__global__ void build_w_kernel(const float* __restrict__ wgt,
                               const int* __restrict__ conn,
                               unsigned short* __restrict__ wdt) {
    int o  = blockIdx.x;
    int ci = threadIdx.x;
    float acc[9];
    #pragma unroll
    for (int t = 0; t < 9; ++t) acc[t] = 0.0f;
    for (int c = 0; c < CPF; ++c) {
        if (conn[o * CPF + c] == ci) {
            #pragma unroll
            for (int t = 0; t < 9; ++t)
                acc[t] += wgt[(o * CPF + c) * 9 + t];
        }
    }
    #pragma unroll
    for (int t = 0; t < 9; ++t)
        wdt[((size_t)(t * COUT + o)) * CIN + ci] = f2bf(acc[t]);
}

// ---------------- Kernel 4: implicit-GEMM conv via bf16 WMMA ----------------
// One wave computes a 32(o) x 32(w) output tile for one (b, h).
// K loop: 9 taps x 8 channel-blocks of 32 -> 72 steps x 4 WMMAs.
// Waves total: NB*HH*(COUT/32)*(64/32) = 4*56*8*2 = 3584; 4 waves/block.
__global__ __launch_bounds__(128)
void conv_wmma_kernel(const __bf16* __restrict__ xt,
                      const __bf16* __restrict__ wdt,
                      float* __restrict__ out) {
    int wave = (blockIdx.x << 2) | (threadIdx.x >> 5);
    int lane = threadIdx.x & 31;
    int l15  = lane & 15;
    int hi   = lane >> 4;

    int wp = wave & 1;            // w half: 0 -> w 0..31, 1 -> w 32..63
    int op = (wave >> 1) & 7;     // o block of 32
    int bh = wave >> 4;
    int h  = bh % HH;
    int b  = bh / HH;

    int obase = op * 32;
    int wbase = wp * 32;

    v8f acc00 = {}, acc01 = {}, acc10 = {}, acc11 = {};

    for (int t = 0; t < 9; ++t) {
        int kh = t / 3, kw = t - 3 * kh;
        // B base: padded row h+kh, column (wbase + l15 + kw), channel-last
        const __bf16* xrow =
            xt + ((size_t)((b * HP + h + kh) * WP + wbase + l15 + kw)) * CIN;
        // A base: tap t, output row obase+l15
        const __bf16* wrow =
            wdt + ((size_t)(t * COUT + obase + l15)) * CIN;

        #pragma unroll
        for (int cb = 0; cb < 8; ++cb) {
            // ---- A fragments (16x32 bf16): lane l15 = row M.
            // elem i -> K = hi*16 + i
            int aoff = cb * 32 + hi * 16;
            v8bf a0lo = *(const v8bf*)(wrow + aoff);
            v8bf a0hi = *(const v8bf*)(wrow + aoff + 8);
            v8bf a1lo = *(const v8bf*)(wrow + 16 * CIN + aoff);
            v8bf a1hi = *(const v8bf*)(wrow + 16 * CIN + aoff + 8);
            v16bf A0 = __builtin_shufflevector(a0lo, a0hi,
                0,1,2,3,4,5,6,7,8,9,10,11,12,13,14,15);
            v16bf A1 = __builtin_shufflevector(a1lo, a1hi,
                0,1,2,3,4,5,6,7,8,9,10,11,12,13,14,15);

            // ---- B fragments (32x16 bf16): lane l15 = column N.
            // elem i -> K = hi*16 + i  => 16 contiguous channels, one 32B load
            int boff = cb * 32 + hi * 16;
            v16bf B0 = *(const v16bf*)(xrow + boff);
            v16bf B1 = *(const v16bf*)(xrow + 16 * CIN + boff);

            acc00 = __builtin_amdgcn_wmma_f32_16x16x32_bf16(
                false, A0, false, B0, (short)0, acc00, false, false);
            acc01 = __builtin_amdgcn_wmma_f32_16x16x32_bf16(
                false, A0, false, B1, (short)0, acc01, false, false);
            acc10 = __builtin_amdgcn_wmma_f32_16x16x32_bf16(
                false, A1, false, B0, (short)0, acc10, false, false);
            acc11 = __builtin_amdgcn_wmma_f32_16x16x32_bf16(
                false, A1, false, B1, (short)0, acc11, false, false);
        }
    }

    // ---- Store D (16x16 f32 per tile): elem r -> row M = hi*8 + r, col = l15
    v8f accs[2][2] = { { acc00, acc01 }, { acc10, acc11 } };
    #pragma unroll
    for (int f = 0; f < 2; ++f) {
        #pragma unroll
        for (int g = 0; g < 2; ++g) {
            int w = wbase + g * 16 + l15;
            if (w < WW) {
                #pragma unroll
                for (int r = 0; r < 8; ++r) {
                    int o = obase + f * 16 + hi * 8 + r;
                    out[(((size_t)b * COUT + o) * HH + h) * WW + w] =
                        accs[f][g][r];
                }
            }
        }
    }
}

// ---------------- Host launcher ---------------------------------------------
extern "C" void kernel_launch(void* const* d_in, const int* in_sizes, int n_in,
                              void* d_out, int out_size, void* d_ws, size_t ws_size,
                              hipStream_t stream) {
    (void)in_sizes; (void)n_in; (void)out_size; (void)ws_size;
    const float* x    = (const float*)d_in[0];
    const float* wgt  = (const float*)d_in[1];
    const int*   conn = (const int*)d_in[2];
    float*       out  = (float*)d_out;

    unsigned short* xt_u  = (unsigned short*)d_ws;
    unsigned short* wdt_u = xt_u + XT_ELEMS;

    // 1) zero padded input region (halo must be 0)
    int ndw = (int)(XT_ELEMS / 2);
    zero32_kernel<<<(ndw + 255) / 256, 256, 0, stream>>>((unsigned int*)d_ws, ndw);

    // 2) repack x -> padded NHWC bf16
    pack_x_kernel<<<dim3(NB * HH, CIN / 32, 2), dim3(32, 8), 0, stream>>>(x, xt_u);

    // 3) densify weights
    build_w_kernel<<<COUT, CIN, 0, stream>>>(wgt, conn, wdt_u);

    // 4) WMMA implicit GEMM
    conv_wmma_kernel<<<(NB * HH * 8 * 2) / 4, 128, 0, stream>>>(
        (const __bf16*)xt_u, (const __bf16*)wdt_u, out);
}
